// Convolution_23708219474701
// MI455X (gfx1250) — matrix operand order repaired and checked by reading, not compiled, not run on hardware
//
#include <hip/hip_runtime.h>

typedef __attribute__((ext_vector_type(16))) _Float16 v16h;
typedef __attribute__((ext_vector_type(8)))  float    v8f;

#define TE      32      // edges per workgroup = 2 WMMA M-tiles
#define WGT     512     // 16 wave32s
#define KDIM    256
#define NDIM    1024
#define WSTRIDE 1028    // padded W-tile row stride (floats): 8*1028 % 64 != 0 -> no bank conflict

__device__ __forceinline__ constexpr float inv_sqrt3() { return 0.57735026918962576f; }

// ---------------------------------------------------------------------------
// Zero the output (harness poisons d_out before timing).
// ---------------------------------------------------------------------------
__global__ void zero_f32(float* __restrict__ p, int n) {
    int i = blockIdx.x * blockDim.x + threadIdx.x;
    if (i < n) p[i] = 0.0f;
}

// ---------------------------------------------------------------------------
// Convert fc_w2 (256 x 1024, f32 row-major) into f16 WMMA B-fragments, with
// the 1/sqrt(256) = 1/16 scale folded in.  B fragment (this code's contiguous k order, same as A):
// 32x16 (KxN) block; lanes 0-15 hold K=0..15 of column N=lane, lanes 16-31
// hold K=16..31; half-slot within lane = K % 16.
// Buffer layout: [n_tile(64)][k_step(8)][lane(32)][half(16)]  -> 512 KB in d_ws.
// ---------------------------------------------------------------------------
__global__ void prep_b_f16(const float* __restrict__ w2, _Float16* __restrict__ b) {
    int idx = blockIdx.x * blockDim.x + threadIdx.x;
    if (idx >= KDIM * NDIM) return;
    int k = idx >> 10;            // 0..255
    int n = idx & 1023;           // 0..1023
    int ks = k >> 5, kk = k & 31;
    int nt = n >> 4, nn = n & 15;
    int lane = nn + 16 * (kk >> 4);
    int slot = kk & 15;
    b[(((nt * 8) + ks) * 32 + lane) * 16 + slot] = (_Float16)(w2[idx] * 0.0625f);
}

// ---------------------------------------------------------------------------
// Fused: h = relu(sc @ w1 / sqrt3); W = h @ B; FCTP; atomic segment-sum.
// Each wave owns BOTH M-tiles x 4 N-tiles, so every B fragment fetched from
// L2 feeds two WMMAs (B traffic: 512 KB per workgroup, fully L2-resident).
// ---------------------------------------------------------------------------
__global__ __launch_bounds__(WGT) void conv_fused(
    const float* __restrict__ nf_g,   const int* __restrict__ src_g,
    const int*   __restrict__ dst_g,  const float* __restrict__ attr_g,
    const float* __restrict__ sc_g,   const float* __restrict__ w1_g,
    const _Float16* __restrict__ Bh,  float* __restrict__ out, int E)
{
    extern __shared__ __align__(16) char smem[];
    float*    Wl    = (float*)smem;                         // TE x WSTRIDE f32 W-tile
    _Float16* Al    = (_Float16*)smem;                      // aliased: A fragments (16 KB), dead after GEMM
    float*    nfL   = (float*)(smem + TE * WSTRIDE * 4);    // TE x 64 gathered node features
    float*    attrL = nfL + TE * 64;                        // TE x 4
    float*    scL   = attrL + TE * 4;                       // TE x 3
    int*      dstL  = (int*)(scL + TE * 3);                 // TE
    int*      srcL  = dstL + TE;                            // TE

    const int tid = threadIdx.x;
    const int e0  = blockIdx.x * TE;

    // ---- stage per-edge small data --------------------------------------
    if (tid < TE) {
        int ee = min(e0 + tid, E - 1);
        srcL[tid] = src_g[ee];
        dstL[tid] = dst_g[ee];
    }
    if (tid < TE * 4) {
        int e = tid >> 2, c = tid & 3;
        int ee = min(e0 + e, E - 1);
        attrL[tid] = attr_g[ee * 4 + c];
    }
    if (tid >= TE * 4 && tid < TE * 4 + TE * 3) {
        int t = tid - TE * 4;
        int e = t / 3, c = t - e * 3;
        int ee = min(e0 + e, E - 1);
        scL[t] = sc_g[ee * 3 + c];
    }
    __syncthreads();

    // ---- gather node features (TE x 64) ---------------------------------
    for (int idx = tid; idx < TE * 64; idx += WGT) {
        int e = idx >> 6, c = idx & 63;
        nfL[idx] = nf_g[srcL[e] * 64 + c];
    }

    // ---- h = relu(sc @ w1 / sqrt3), written as f16 A-fragments ----------
    // A 16x32 fragments, contiguous k order (same as B): K 0-15 -> lanes 0-15 slots 0-15;
    // K 16-31 -> lanes 16-31 slots 0-15.
    for (int idx = tid; idx < TE * KDIM; idx += WGT) {
        int e = idx >> 8, k = idx & 255;
        float hv = scL[e * 3 + 0] * w1_g[k]
                 + scL[e * 3 + 1] * w1_g[KDIM + k]
                 + scL[e * 3 + 2] * w1_g[2 * KDIM + k];
        hv = fmaxf(hv * inv_sqrt3(), 0.0f);
        int mt = e >> 4, er = e & 15, ks = k >> 5, kk = k & 31;
        int lane = er + 16 * (kk >> 4);
        int slot = kk & 15;
        Al[((mt * 8 + ks) * 32 + lane) * 16 + slot] = (_Float16)hv;
    }
    __syncthreads();

    // ---- GEMM: W(32x1024) += A(32x256) x B(256x1024), f32 accum ---------
    // wave w: M-tiles {0,1} x N-tiles [4w, 4w+4); each B fragment -> 2 WMMAs.
    const int wv   = tid >> 5;           // wave 0..15
    const int lane = tid & 31;
    const int nb   = wv * 4;             // first of 4 N-tiles for this wave

    v8f acc0[4] = {};                    // M-tile 0 accumulators
    v8f acc1[4] = {};                    // M-tile 1 accumulators
#pragma unroll
    for (int ks = 0; ks < 8; ++ks) {
        v16h a0 = *(const v16h*)&Al[(ks * 32 + lane) * 16];
        v16h a1 = *(const v16h*)&Al[((8 + ks) * 32 + lane) * 16];
        if (ks < 7)
            __builtin_prefetch(&Bh[((nb * 8 + ks + 1) * 32 + lane) * 16], 0, 1);
#pragma unroll
        for (int nt = 0; nt < 4; ++nt) {
            v16h b = *(const v16h*)&Bh[(((nb + nt) * 8 + ks) * 32 + lane) * 16];
            acc0[nt] = __builtin_amdgcn_wmma_f32_16x16x32_f16(
                false, a0, false, b, (short)0, acc0[nt], false, false);
            acc1[nt] = __builtin_amdgcn_wmma_f32_16x16x32_f16(
                false, a1, false, b, (short)0, acc1[nt], false, false);
        }
    }
    __syncthreads();   // everyone done reading Al before W overwrites it

    // ---- spill accumulators to LDS W-tile -------------------------------
    // C/D layout: VGPR r, lanes 0-15 -> M=r, lanes 16-31 -> M=r+8; N = lane%16.
    {
        const int mrow = 8 * (lane >> 4);
        const int ncol = lane & 15;
#pragma unroll
        for (int nt = 0; nt < 4; ++nt) {
            int j = (nb + nt) * 16 + ncol;
#pragma unroll
            for (int r = 0; r < 8; ++r) {
                Wl[(mrow + r) * WSTRIDE + j]      = acc0[nt][r];
                Wl[(16 + mrow + r) * WSTRIDE + j] = acc1[nt][r];
            }
        }
    }
    __syncthreads();

    // ---- FCTP + atomic scatter: one (edge, o) pair per thread -----------
    const int e = tid >> 4;      // 0..31
    const int o = tid & 15;      // 0..15
    const float sh_s = attrL[e * 4 + 0];
    const float sv0  = attrL[e * 4 + 1];
    const float sv1  = attrL[e * 4 + 2];
    const float sv2  = attrL[e * 4 + 3];
    const float* nfe = &nfL[e * 64];
    const float* We  = &Wl[e * WSTRIDE];

    float os = 0.0f, a3 = 0.0f, o0 = 0.0f, o1 = 0.0f, o2 = 0.0f;
#pragma unroll
    for (int i = 0; i < 16; ++i) {
        float si = nfe[i];
        float vx = nfe[16 + i * 3 + 0];
        float vy = nfe[16 + i * 3 + 1];
        float vz = nfe[16 + i * 3 + 2];
        float dt = (vx * sv0 + vy * sv1 + vz * sv2) * inv_sqrt3();
        os += We[i * 16 + o] * (si * sh_s) + We[256 + i * 16 + o] * dt;
        float w3 = We[512 + i * 16 + o];
        float w4 = We[768 + i * 16 + o];
        a3 += w3 * si;
        o0 += w4 * vx; o1 += w4 * vy; o2 += w4 * vz;
    }
    // 1/sqrt(2*MUL) * 1/sqrt(NUM_NEIGHBORS) = 1/sqrt(32) * 0.25
    const float fs = 0.17677669529663687f * 0.25f;
    if (e0 + e < E) {
        float* orow = &out[dstL[e] * 64];
        atomicAdd(&orow[o],                os * fs);
        atomicAdd(&orow[16 + o * 3 + 0], (a3 * sv0 + o0 * sh_s) * fs);
        atomicAdd(&orow[16 + o * 3 + 1], (a3 * sv1 + o1 * sh_s) * fs);
        atomicAdd(&orow[16 + o * 3 + 2], (a3 * sv2 + o2 * sh_s) * fs);
    }
}

// ---------------------------------------------------------------------------
extern "C" void kernel_launch(void* const* d_in, const int* in_sizes, int n_in,
                              void* d_out, int out_size, void* d_ws, size_t ws_size,
                              hipStream_t stream) {
    const float* nf   = (const float*)d_in[0];   // (20000, 64)
    const int*   src  = (const int*)  d_in[1];   // (E,)
    const int*   dst  = (const int*)  d_in[2];   // (E,)
    const float* attr = (const float*)d_in[3];   // (E, 4)
    const float* sc   = (const float*)d_in[4];   // (E, 3)
    const float* w1   = (const float*)d_in[5];   // (3, 256)
    const float* w2   = (const float*)d_in[6];   // (256, 1024)
    float* out = (float*)d_out;
    _Float16* Bh = (_Float16*)d_ws;              // 512 KB f16 B-fragments
    const int E = in_sizes[1];

    zero_f32<<<(out_size + 511) / 512, 512, 0, stream>>>(out, out_size);
    prep_b_f16<<<(KDIM * NDIM + 255) / 256, 256, 0, stream>>>(w2, Bh);

    const size_t shbytes = (size_t)(TE * WSTRIDE + TE * 64 + TE * 4 + TE * 3) * 4
                         + (size_t)(2 * TE) * 4;
    conv_fused<<<(E + TE - 1) / TE, WGT, shbytes, stream>>>(
        nf, src, dst, attr, sc, w1, Bh, out, E);
}
